// ProteinMPNN_pLDDT_16913581211862
// MI455X (gfx1250) — compile-verified, matrix operand order checked by reading, not run on hardware
//
#include <hip/hip_runtime.h>
#include <hip/hip_bf16.h>
#include <math.h>

#define HD   128
#define KN   48
#define NB   2048
#define BBATCH 4
#define BN   (BBATCH*NB)       // 8192 nodes
#define SCALE_INV (1.0f/30.0f)
#define LN_EPS 1e-5f

typedef __attribute__((ext_vector_type(16))) _Float16 v16h;
typedef __attribute__((ext_vector_type(8)))  float    v8f;

// Branch-free erf-GELU: Abramowitz–Stegun 7.1.26 (|err| < 1.5e-7) with
// hardware rcp/exp2 — no EXEC divergence, co-issues with WMMA (XDL=TRANS).
__device__ __forceinline__ float gelu_erf(float x){
  float s  = x * 0.70710678118654752440f;     // x / sqrt(2)
  float ax = __builtin_fabsf(s);
  float t  = __builtin_amdgcn_rcpf(__builtin_fmaf(0.3275911f, ax, 1.0f));
  float p  = __builtin_fmaf(1.061405429f, t, -1.453152027f);
  p = __builtin_fmaf(p, t, 1.421413741f);
  p = __builtin_fmaf(p, t, -0.284496736f);
  p = __builtin_fmaf(p, t, 0.254829592f);
  p = p * t;
  float e  = __builtin_amdgcn_exp2f(-ax * ax * 1.4426950408889634f);
  float er = __builtin_copysignf(1.0f - p * e, s);
  return 0.5f * x * (1.0f + er);
}

__device__ __forceinline__ float wred32(float v){
  #pragma unroll
  for (int off = 16; off > 0; off >>= 1) v += __shfl_xor(v, off, 32);
  return v;
}

// ---------------------------------------------------------------------------
// A-fragment loader: 16x32 f16 tile from a row-major f16 LDS buffer.
// ISA 7.12.2 (16-bit A 16x32): lanes 0-15 hold row=lane, K={0..7,16..23};
// lanes 16-31 hold row=lane-16, K={8..15,24..31}.
// ---------------------------------------------------------------------------
__device__ __forceinline__ v16h load_a_f16(const _Float16* src, int ld, int kt, int lane){
  const int row = lane & 15;
  const int kb  = kt * 32 + ((lane & 16) ? 8 : 0);
  const _Float16* p = src + row * ld + kb;
  v16h a;
  #pragma unroll
  for (int i = 0; i < 8; ++i){ a[i] = p[i]; a[i+8] = p[16+i]; }
  return a;
}

// One 16x16 output tile: C = A(16 x 32*ktiles) * W(32*ktiles x 16) + bias.
// Weights pre-packed into B-fragment order: frag (kt,nt), 32 lanes x 16 f16.
__device__ __forceinline__ v8f gemm_tile(const _Float16* A, int ldA, int ktiles,
                                         const _Float16* Wp, int Nt, int nt,
                                         float bias_v, int lane){
  v8f acc;
  #pragma unroll
  for (int v = 0; v < 8; ++v) acc[v] = bias_v;
  #pragma unroll 4
  for (int kt = 0; kt < ktiles; ++kt){
    v16h a = load_a_f16(A, ldA, kt, lane);
    v16h b = *(const v16h*)(Wp + (((size_t)kt * Nt + nt) * 32 + lane) * 16);
    acc = __builtin_amdgcn_wmma_f32_16x16x32_f16(false, a, false, b,
                                                 (short)0, acc, false, false);
  }
  return acc;
}

// ---------------------------------------------------------------------------
// Weight repack: [din,dout] f32 row-major -> f16 B-fragments (mirror of A map
// with N playing the role of M: lane holds col=lane&15, same K interleave).
// ---------------------------------------------------------------------------
__global__ void pack_w_kernel(const float* __restrict__ W, _Float16* __restrict__ dst,
                              int din, int dout){
  int id = blockIdx.x * blockDim.x + threadIdx.x;
  int total = din * dout;
  if (id >= total) return;
  int Nt   = dout >> 4;
  int frag = id >> 9;          // /512 f16 per fragment
  int rem  = id & 511;
  int lane = rem >> 4;
  int i    = rem & 15;
  int kt = frag / Nt, nt = frag % Nt;
  int kb = (lane & 16) ? 8 : 0;
  int k  = kt * 32 + ((i < 8) ? (kb + i) : (kb + 16 + i - 8));
  int col = nt * 16 + (lane & 15);
  dst[id] = (_Float16)W[(size_t)k * dout + col];
}

// ---------------------------------------------------------------------------
// Node message + LN1: one workgroup per node (48 edges = 3 tiles of 16 rows).
// ---------------------------------------------------------------------------
__global__ __launch_bounds__(128)
void node_msg_kernel(const float* __restrict__ hV, const float* __restrict__ hE,
                     const int* __restrict__ Eidx, const float* __restrict__ mask_att,
                     const float* __restrict__ conf,
                     const _Float16* __restrict__ W1p, const float* __restrict__ b1,
                     const _Float16* __restrict__ W2p, const float* __restrict__ b2,
                     const _Float16* __restrict__ W3p, const float* __restrict__ b3,
                     const float* __restrict__ ln1g, const float* __restrict__ ln1b,
                     float* __restrict__ hV1)
{
  __shared__ _Float16 Xs[KN * 384];        // 36 KB : [hV_i | hE | hV_j] f16
  __shared__ _Float16 Hb[3][16 * 128];     // 12 KB : layer-1 gelu out, per wave
  __shared__ _Float16 Gb[3][16 * 128];     // 12 KB : layer-2 gelu out, per wave
  __shared__ float    s_scale[KN];
  __shared__ int      sidx[KN];
  __shared__ float    dh[HD];
  __shared__ float    red[4];

  const int node = blockIdx.x;
  const int bidx = node / NB;
  const int t    = threadIdx.x;            // 0..127 == channel
  const int lane = t & 31, wid = t >> 5;
  const int l15  = lane & 15, h16 = lane >> 4;

  if (t < KN){
    sidx[t]    = Eidx[node * KN + t];
    s_scale[t] = mask_att[node * KN + t] * conf[node * KN + t];
  }
  dh[t] = 0.0f;
  __syncthreads();

  const float* hVc = hV + (size_t)node * HD;
  {
    float cv = hVc[t];
    for (int k = 0; k < KN; ++k){
      Xs[k * 384 + t]        = (_Float16)cv;
      Xs[k * 384 + 128 + t]  = (_Float16)hE[((size_t)node * KN + k) * HD + t];
      Xs[k * 384 + 256 + t]  = (_Float16)hV[((size_t)bidx * NB + sidx[k]) * HD + t];
    }
  }
  __syncthreads();

  if (wid < 3){
    const _Float16* Xw = Xs + wid * 16 * 384;
    _Float16* H1 = Hb[wid];
    _Float16* H2 = Gb[wid];
    // layer 1: (16x384) @ (384x128)
    for (int nt = 0; nt < 8; ++nt){
      v8f acc = gemm_tile(Xw, 384, 12, W1p, 8, nt, b1[nt*16 + l15], lane);
      #pragma unroll
      for (int v = 0; v < 8; ++v){
        int r = v + 8 * h16;
        H1[r * 128 + nt * 16 + l15] = (_Float16)gelu_erf(acc[v]);
      }
    }
    // layer 2: (16x128) @ (128x128)
    for (int nt = 0; nt < 8; ++nt){
      v8f acc = gemm_tile(H1, 128, 4, W2p, 8, nt, b2[nt*16 + l15], lane);
      #pragma unroll
      for (int v = 0; v < 8; ++v){
        int r = v + 8 * h16;
        H2[r * 128 + nt * 16 + l15] = (_Float16)gelu_erf(acc[v]);
      }
    }
    // layer 3 + masked reduction over K.  Hoist the per-row scales out of the
    // nt loop (LDS atomics below otherwise block the compiler from hoisting).
    float sc[8];
    #pragma unroll
    for (int v = 0; v < 8; ++v) sc[v] = s_scale[wid * 16 + v + 8 * h16];
    for (int nt = 0; nt < 8; ++nt){
      v8f acc = gemm_tile(H2, 128, 4, W3p, 8, nt, b3[nt*16 + l15], lane);
      float part = 0.0f;
      #pragma unroll
      for (int v = 0; v < 8; ++v) part += acc[v] * sc[v];
      atomicAdd(&dh[nt * 16 + l15], part);
    }
  }
  __syncthreads();

  // LN1 over 128 channels (block-wide)
  float x = hVc[t] + dh[t] * SCALE_INV;
  float s = wred32(x);
  if (lane == 0) red[wid] = s;
  __syncthreads();
  float mean = (red[0] + red[1] + red[2] + red[3]) * (1.0f / 128.0f);
  float d = x - mean;
  __syncthreads();
  float s2 = wred32(d * d);
  if (lane == 0) red[wid] = s2;
  __syncthreads();
  float var = (red[0] + red[1] + red[2] + red[3]) * (1.0f / 128.0f);
  hV1[(size_t)node * HD + t] = d * rsqrtf(var + LN_EPS) * ln1g[t] + ln1b[t];
}

// ---------------------------------------------------------------------------
// FFN + LN2 + mask_V: one wave per 16-node tile (64 nodes per workgroup).
// ---------------------------------------------------------------------------
__global__ __launch_bounds__(128)
void ffn_kernel(const float* __restrict__ hV1,
                const _Float16* __restrict__ Winp, const float* __restrict__ binn,
                const _Float16* __restrict__ Woutp, const float* __restrict__ bout,
                const float* __restrict__ ln2g, const float* __restrict__ ln2b,
                const float* __restrict__ maskV,
                float* __restrict__ hVout)
{
  __shared__ _Float16 X1[4][16 * 128];   // 16 KB
  __shared__ _Float16 Hm[4][16 * 512];   // 64 KB
  __shared__ float    Dm[4][16 * 128];   // 32 KB

  const int t = threadIdx.x, lane = t & 31, wid = t >> 5;
  const int l15 = lane & 15, h16 = lane >> 4;
  const int node0 = blockIdx.x * 64 + wid * 16;

  _Float16* Xw = X1[wid];
  _Float16* Hw = Hm[wid];
  float*    Dw = Dm[wid];

  for (int r = 0; r < 16; ++r){
    const float* src = hV1 + (size_t)(node0 + r) * HD;
    #pragma unroll
    for (int q = 0; q < 4; ++q){
      int c = lane + 32 * q;
      Xw[r * 128 + c] = (_Float16)src[c];
    }
  }
  // GEMM1: (16x128) @ (128x512), erf-gelu
  for (int nt = 0; nt < 32; ++nt){
    v8f acc = gemm_tile(Xw, 128, 4, Winp, 32, nt, binn[nt*16 + l15], lane);
    #pragma unroll
    for (int v = 0; v < 8; ++v){
      int r = v + 8 * h16;
      Hw[r * 512 + nt * 16 + l15] = (_Float16)gelu_erf(acc[v]);
    }
  }
  // GEMM2: (16x512) @ (512x128)
  for (int nt = 0; nt < 8; ++nt){
    v8f acc = gemm_tile(Hw, 512, 16, Woutp, 8, nt, bout[nt*16 + l15], lane);
    #pragma unroll
    for (int v = 0; v < 8; ++v){
      int r = v + 8 * h16;
      Dw[r * 128 + nt * 16 + l15] = acc[v];
    }
  }
  // per-row residual + LN2 + mask_V
  for (int r = 0; r < 16; ++r){
    int bn = node0 + r;
    const float* res = hV1 + (size_t)bn * HD;
    float xs[4]; float s = 0.0f;
    #pragma unroll
    for (int q = 0; q < 4; ++q){
      int c = lane + 32 * q;
      xs[q] = res[c] + Dw[r * 128 + c];
      s += xs[q];
    }
    float mean = wred32(s) * (1.0f / 128.0f);
    float vs = 0.0f;
    #pragma unroll
    for (int q = 0; q < 4; ++q){ float dd = xs[q] - mean; vs += dd * dd; }
    float inv = rsqrtf(wred32(vs) * (1.0f / 128.0f) + LN_EPS);
    float mk = maskV[bn];
    #pragma unroll
    for (int q = 0; q < 4; ++q){
      int c = lane + 32 * q;
      hVout[(size_t)bn * HD + c] = ((xs[q] - mean) * inv * ln2g[c] + ln2b[c]) * mk;
    }
  }
}

// ---------------------------------------------------------------------------
// Edge update + LN3: one workgroup per node; uses post-FFN h_V for gathers.
// ---------------------------------------------------------------------------
__global__ __launch_bounds__(128)
void edge_kernel(const float* __restrict__ hVf, const float* __restrict__ hE,
                 const int* __restrict__ Eidx,
                 const _Float16* __restrict__ W11p, const float* __restrict__ b11,
                 const _Float16* __restrict__ W12p, const float* __restrict__ b12,
                 const _Float16* __restrict__ W13p, const float* __restrict__ b13,
                 const float* __restrict__ ln3g, const float* __restrict__ ln3b,
                 float* __restrict__ hEout)
{
  __shared__ _Float16 Xs[KN * 384];
  __shared__ _Float16 Hb[3][16 * 128];
  __shared__ _Float16 Gb[3][16 * 128];
  __shared__ float    Ms[3][16 * 128];   // layer-3 message, f32 for LN
  __shared__ int      sidx[KN];

  const int node = blockIdx.x;
  const int bidx = node / NB;
  const int t    = threadIdx.x;
  const int lane = t & 31, wid = t >> 5;
  const int l15  = lane & 15, h16 = lane >> 4;

  if (t < KN) sidx[t] = Eidx[node * KN + t];
  __syncthreads();

  {
    float cv = hVf[(size_t)node * HD + t];
    for (int k = 0; k < KN; ++k){
      Xs[k * 384 + t]       = (_Float16)cv;
      Xs[k * 384 + 128 + t] = (_Float16)hE[((size_t)node * KN + k) * HD + t];
      Xs[k * 384 + 256 + t] = (_Float16)hVf[((size_t)bidx * NB + sidx[k]) * HD + t];
    }
  }
  __syncthreads();

  if (wid < 3){
    const _Float16* Xw = Xs + wid * 16 * 384;
    _Float16* H1 = Hb[wid];
    _Float16* H2 = Gb[wid];
    float*    Mw = Ms[wid];
    for (int nt = 0; nt < 8; ++nt){
      v8f acc = gemm_tile(Xw, 384, 12, W11p, 8, nt, b11[nt*16 + l15], lane);
      #pragma unroll
      for (int v = 0; v < 8; ++v){
        int r = v + 8 * h16;
        H1[r * 128 + nt * 16 + l15] = (_Float16)gelu_erf(acc[v]);
      }
    }
    for (int nt = 0; nt < 8; ++nt){
      v8f acc = gemm_tile(H1, 128, 4, W12p, 8, nt, b12[nt*16 + l15], lane);
      #pragma unroll
      for (int v = 0; v < 8; ++v){
        int r = v + 8 * h16;
        H2[r * 128 + nt * 16 + l15] = (_Float16)gelu_erf(acc[v]);
      }
    }
    for (int nt = 0; nt < 8; ++nt){
      v8f acc = gemm_tile(H2, 128, 4, W13p, 8, nt, b13[nt*16 + l15], lane);
      #pragma unroll
      for (int v = 0; v < 8; ++v){
        int r = v + 8 * h16;
        Mw[r * 128 + nt * 16 + l15] = acc[v];
      }
    }
    // per-row residual + LN3, write h_E out
    for (int r = 0; r < 16; ++r){
      int k = wid * 16 + r;
      size_t eo = ((size_t)node * KN + k) * HD;
      float xs[4]; float s = 0.0f;
      #pragma unroll
      for (int q = 0; q < 4; ++q){
        int c = lane + 32 * q;
        xs[q] = hE[eo + c] + Mw[r * 128 + c];
        s += xs[q];
      }
      float mean = wred32(s) * (1.0f / 128.0f);
      float vs = 0.0f;
      #pragma unroll
      for (int q = 0; q < 4; ++q){ float dd = xs[q] - mean; vs += dd * dd; }
      float inv = rsqrtf(wred32(vs) * (1.0f / 128.0f) + LN_EPS);
      #pragma unroll
      for (int q = 0; q < 4; ++q){
        int c = lane + 32 * q;
        hEout[eo + c] = (xs[q] - mean) * inv * ln3g[c] + ln3b[c];
      }
    }
  }
}

// ---------------------------------------------------------------------------
extern "C" void kernel_launch(void* const* d_in, const int* in_sizes, int n_in,
                              void* d_out, int out_size, void* d_ws, size_t ws_size,
                              hipStream_t stream) {
  const float* hV    = (const float*)d_in[0];
  const float* hE    = (const float*)d_in[1];
  const int*   Eidx  = (const int*)d_in[2];
  const float* maskV = (const float*)d_in[3];
  const float* maskA = (const float*)d_in[4];
  const float* conf  = (const float*)d_in[5];
  const float* W1  = (const float*)d_in[6];  const float* b1  = (const float*)d_in[7];
  const float* W2  = (const float*)d_in[8];  const float* b2  = (const float*)d_in[9];
  const float* W3  = (const float*)d_in[10]; const float* b3  = (const float*)d_in[11];
  const float* W11 = (const float*)d_in[12]; const float* b11 = (const float*)d_in[13];
  const float* W12 = (const float*)d_in[14]; const float* b12 = (const float*)d_in[15];
  const float* W13 = (const float*)d_in[16]; const float* b13 = (const float*)d_in[17];
  const float* Win = (const float*)d_in[18]; const float* binn= (const float*)d_in[19];
  const float* Wout= (const float*)d_in[20]; const float* bout= (const float*)d_in[21];
  const float* ln1g= (const float*)d_in[22]; const float* ln1b= (const float*)d_in[23];
  const float* ln2g= (const float*)d_in[24]; const float* ln2b= (const float*)d_in[25];
  const float* ln3g= (const float*)d_in[26]; const float* ln3b= (const float*)d_in[27];

  _Float16* wsf = (_Float16*)d_ws;
  size_t o = 0;
  _Float16* W1p  = wsf + o; o += 384*128;
  _Float16* W2p  = wsf + o; o += 128*128;
  _Float16* W3p  = wsf + o; o += 128*128;
  _Float16* W11p = wsf + o; o += 384*128;
  _Float16* W12p = wsf + o; o += 128*128;
  _Float16* W13p = wsf + o; o += 128*128;
  _Float16* Winp = wsf + o; o += 128*512;
  _Float16* Woutp= wsf + o; o += 512*128;
  float* hV1 = (float*)(wsf + o);          // 8192*128 f32 intermediate

  auto pack = [&](const float* W, _Float16* dst, int din, int dout){
    int total = din * dout;
    hipLaunchKernelGGL(pack_w_kernel, dim3((total + 255) / 256), dim3(256), 0, stream,
                       W, dst, din, dout);
  };
  pack(W1,  W1p,  384, 128);
  pack(W2,  W2p,  128, 128);
  pack(W3,  W3p,  128, 128);
  pack(W11, W11p, 384, 128);
  pack(W12, W12p, 128, 128);
  pack(W13, W13p, 128, 128);
  pack(Win, Winp, 128, 512);
  pack(Wout,Woutp,512, 128);

  float* outV = (float*)d_out;
  float* outE = outV + (size_t)BN * HD;

  hipLaunchKernelGGL(node_msg_kernel, dim3(BN), dim3(128), 0, stream,
                     hV, hE, Eidx, maskA, conf,
                     W1p, b1, W2p, b2, W3p, b3, ln1g, ln1b, hV1);

  hipLaunchKernelGGL(ffn_kernel, dim3(BN / 64), dim3(128), 0, stream,
                     hV1, Winp, binn, Woutp, bout, ln2g, ln2b, maskV, outV);

  hipLaunchKernelGGL(edge_kernel, dim3(BN), dim3(128), 0, stream,
                     outV, hE, Eidx,
                     W11p, b11, W12p, b12, W13p, b13, ln3g, ln3b, outE);
}
